// SequenceMultiHeadAttention_23313082482837
// MI455X (gfx1250) — matrix operand order repaired and checked by reading, not compiled, not run on hardware
//
#include <hip/hip_runtime.h>
#include <hip/hip_bf16.h>

// ---------------------------------------------------------------------------
// Types for CDNA5 WMMA (wave32): v_wmma_f32_16x16x32_bf16
// ---------------------------------------------------------------------------
typedef __attribute__((ext_vector_type(16))) __bf16 bf16x16;
typedef __attribute__((ext_vector_type(8)))  __bf16 bf16x8;
typedef __attribute__((ext_vector_type(4)))  __bf16 bf16x4;
typedef __attribute__((ext_vector_type(8)))  float  f32x8;
typedef __attribute__((ext_vector_type(4)))  float  f32x4;

#define T_TOK 4096
#define F_IN  512
#define NHEAD 8
#define ADIM  128
#define HIDD  128
#define NCOLS 1024   // H*A == H*HID
#define F_OUT 512
#define SCALE 0.088388347648318447f   // 128^-0.5

// Hardware BF16 convert (RNE on gfx1250) instead of manual bit-twiddling.
static __device__ __forceinline__ __bf16 f2bf(float f) { return (__bf16)f; }

static __device__ __forceinline__ bf16x16 combine8(bf16x8 a, bf16x8 b) {
    bf16x16 r;
#pragma unroll
    for (int i = 0; i < 8; ++i) { r[i] = a[i]; r[i + 8] = b[i]; }
    return r;
}

// A-fragment (16x32 bf16, M x K): lane holds row = lane&15,
// K elements (lane>>4)*8 + {0..7} and (lane>>4)*8 + 16 + {0..7}.
// p must point at  row*ld + (lane>>4)*8  of a row-major [M][K>=32] array.
static __device__ __forceinline__ bf16x16 load_afrag(const __bf16* p) {
    bf16x8 c0 = *(const bf16x8*)(p);
    bf16x8 c1 = *(const bf16x8*)(p + 16);
    return combine8(c0, c1);
}

static __device__ __forceinline__ f32x8 wmma_bf16(bf16x16 a, bf16x16 b, f32x8 c) {
    return __builtin_amdgcn_wmma_f32_16x16x32_bf16(
        /*neg_a=*/false, a, /*neg_b=*/false, b,
        /*c_mod=*/(short)0, c, /*reuse_a=*/false, /*reuse_b=*/false);
}

// ---------------------------------------------------------------------------
// Kernel 1: projection GEMM  Out[bf16] = bf16(X[f32] @ W[f32])
//   X: [4096,512] f32, W: [512,1024] f32.
//   TR==false -> Out row-major [4096,1024] bf16 (Q,K)
//   TR==true  -> Out col-major [1024,4096] bf16 (V transposed for PV stage)
// Block: 256 thr (8 waves, 2x4), tile 128x128, K step 32.
// ---------------------------------------------------------------------------
template <bool TR>
__global__ __launch_bounds__(256)
void proj_gemm(const float* __restrict__ X, const float* __restrict__ W,
               __bf16* __restrict__ Out) {
    __shared__ __align__(32) __bf16 Xl[128 * 32];   // [row][k]
    __shared__ __align__(32) __bf16 Wl[128 * 32];   // [n][k] (transposed)

    const int tid  = threadIdx.x;
    const int lane = tid & 31;
    const int wv   = tid >> 5;
    const int hi16 = lane >> 4;
    const int tc   = lane & 15;
    const int wm   = (wv >> 2) * 64;   // wave M offset (0,64)
    const int wn   = (wv & 3) * 32;    // wave N offset (0,32,64,96)
    const int m0   = blockIdx.x * 128;
    const int n0   = blockIdx.y * 128;

    f32x8 acc[4][2] = {};

    for (int k0 = 0; k0 < F_IN; k0 += 32) {
        // ---- cooperative fill: 128x32 X tile (row-major) -----------------
#pragma unroll
        for (int i = 0; i < 4; ++i) {
            int lin = tid * 4 + i * 1024;
            int r = lin >> 5, c = lin & 31;
            f32x4 xv = *(const f32x4*)(X + (size_t)(m0 + r) * F_IN + k0 + c);
            bf16x4 bv;
#pragma unroll
            for (int e = 0; e < 4; ++e) bv[e] = f2bf(xv[e]);
            *(bf16x4*)(Xl + lin) = bv;
        }
        // ---- cooperative fill: 32x128 W tile, stored transposed [n][k] ---
#pragma unroll
        for (int i = 0; i < 4; ++i) {
            int lin = tid * 4 + i * 1024;
            int wk = lin >> 7, nn = lin & 127;
            f32x4 wvv = *(const f32x4*)(W + (size_t)(k0 + wk) * NCOLS + n0 + nn);
#pragma unroll
            for (int e = 0; e < 4; ++e) Wl[(nn + e) * 32 + wk] = f2bf(wvv[e]);
        }
        __syncthreads();

        bf16x16 bf[2];
#pragma unroll
        for (int j = 0; j < 2; ++j)
            bf[j] = load_afrag(Wl + (wn + j * 16 + tc) * 32 + hi16 * 8);
#pragma unroll
        for (int i = 0; i < 4; ++i) {
            bf16x16 af = load_afrag(Xl + (wm + i * 16 + tc) * 32 + hi16 * 8);
#pragma unroll
            for (int j = 0; j < 2; ++j)
                acc[i][j] = wmma_bf16(af, bf[j], acc[i][j]);
        }
        __syncthreads();
    }

#pragma unroll
    for (int i = 0; i < 4; ++i)
#pragma unroll
        for (int j = 0; j < 2; ++j)
#pragma unroll
            for (int v = 0; v < 8; ++v) {
                int row = m0 + wm + i * 16 + v + 8 * hi16;
                int col = n0 + wn + j * 16 + tc;
                __bf16 val = f2bf(acc[i][j][v]);
                if (TR) Out[(size_t)col * T_TOK + row] = val;
                else    Out[(size_t)row * NCOLS + col] = val;
            }
}

// ---------------------------------------------------------------------------
// Kernel 2: segmented flash attention. One wave per (16-query tile, head).
//   Q,K : [T, H*A] bf16 row-major;  Vt : [H*HID, T] bf16;  idx: [T] sorted.
//   Res : [T, H*HID] bf16.
// Scores S (M=s keys, N=t queries): softmax over M = per-lane 8 rows +
// shfl_xor(16).  P transposed C-layout -> A-layout through per-wave LDS.
// No block-level barriers inside the key loop (waves diverge per segment).
// ---------------------------------------------------------------------------
__global__ __launch_bounds__(128)
void attn_kernel(const __bf16* __restrict__ Q, const __bf16* __restrict__ Kb,
                 const __bf16* __restrict__ Vt, const int* __restrict__ idx,
                 __bf16* __restrict__ Res) {
    __shared__ __align__(32) __bf16 pA[4][16 * 32];  // per-wave P tile [t][s]
    __shared__ __align__(32) float  sB[4][16];       // per-wave row factors

    const int tid   = threadIdx.x;
    const int slot  = tid >> 5;
    const int lane  = tid & 31;
    const int hi16  = lane >> 4;
    const int tc    = lane & 15;
    const int wid   = blockIdx.x * 4 + slot;
    const int qt    = wid >> 3;          // 0..255
    const int h     = wid & 7;
    const int t0    = qt * 16;
    const int tg    = t0 + tc;

    __bf16* myP = pA[slot];
    float*  myS = sB[slot];

    // per-column segment bounds via binary search over sorted idx
    const int myIdx = idx[tg];
    int lo = 0, hb = T_TOK;
    while (lo < hb) { int mid = (lo + hb) >> 1; if (idx[mid] < myIdx) lo = mid + 1; else hb = mid; }
    const int lo_t = lo;
    int lo2 = 0, hb2 = T_TOK;
    while (lo2 < hb2) { int mid = (lo2 + hb2) >> 1; if (idx[mid] <= myIdx) lo2 = mid + 1; else hb2 = mid; }
    const int hi_t = lo2;
    const int loAll = __shfl(lo_t, 0, 32);
    const int hiAll = __shfl(hi_t, 15, 32);

    // Q fragments (B layout: n = t = lane&15, k = (lane>>4)*8 + {0..7}, +16 + {0..7})
    bf16x16 qf[4];
    const __bf16* qp = Q + (size_t)tg * NCOLS + h * ADIM + hi16 * 8;
#pragma unroll
    for (int ks = 0; ks < 4; ++ks) qf[ks] = load_afrag(qp + ks * 32);

    f32x8 O[8] = {};
    float m = -1e30f, l = 0.f;

    for (int sBase = loAll & ~31; sBase < hiAll; sBase += 32) {
        // prefetch next key tile (K rows + V stream) into caches
        if (sBase + 32 < hiAll) {
            __builtin_prefetch(Kb + (size_t)(sBase + 32 + tc) * NCOLS + h * ADIM + hi16 * 8, 0, 3);
            __builtin_prefetch(Vt + (size_t)(h * HIDD + lane * 4) * T_TOK + sBase + 32, 0, 3);
        }
        float pf[2][8];
        float tmax = -1e30f;
#pragma unroll
        for (int st = 0; st < 2; ++st) {
            f32x8 acc = {};
            const int srow = sBase + st * 16 + tc;
            const __bf16* kp = Kb + (size_t)srow * NCOLS + h * ADIM + hi16 * 8;
#pragma unroll
            for (int ks = 0; ks < 4; ++ks)
                acc = wmma_bf16(load_afrag(kp + ks * 32), qf[ks], acc);
#pragma unroll
            for (int v = 0; v < 8; ++v) {
                int s = sBase + st * 16 + v + 8 * hi16;
                float val = (s >= lo_t && s < hi_t) ? acc[v] * SCALE : -1e30f;
                pf[st][v] = val;
                tmax = fmaxf(tmax, val);
            }
        }
        tmax = fmaxf(tmax, __shfl_xor(tmax, 16, 32));
        const float mNew  = fmaxf(m, tmax);
        const float alpha = __expf(m - mNew);
        m = mNew;
        float sum = 0.f;
#pragma unroll
        for (int st = 0; st < 2; ++st)
#pragma unroll
            for (int v = 0; v < 8; ++v) { pf[st][v] = __expf(pf[st][v] - mNew); sum += pf[st][v]; }
        sum += __shfl_xor(sum, 16, 32);
        l = l * alpha + sum;

        // C-layout -> A-layout transpose of P through LDS: pA[t][s]
#pragma unroll
        for (int st = 0; st < 2; ++st) {
            bf16x8 pk;
#pragma unroll
            for (int v = 0; v < 8; ++v) pk[v] = f2bf(pf[st][v]);
            *(bf16x8*)(myP + tc * 32 + st * 16 + hi16 * 8) = pk;
        }
        if (lane < 16) myS[tc] = alpha;          // per-row (t) rescale factors
        __threadfence_block();                    // same-wave DS ordering

        f32x8 aRow = *(const f32x8*)(myS + hi16 * 8);
        bf16x16 pfrag = load_afrag(myP + tc * 32 + hi16 * 8);
#pragma unroll
        for (int dt = 0; dt < 8; ++dt) {
            const __bf16* vp = Vt + (size_t)(h * HIDD + dt * 16 + tc) * T_TOK
                             + sBase + hi16 * 8;
            bf16x16 vf = load_afrag(vp);    // B layout: k = s in table order
            f32x8 o = O[dt] * aRow;
            O[dt] = wmma_bf16(pfrag, vf, o);
        }
        __threadfence_block();
    }

    if (lane < 16) myS[tc] = l;
    __threadfence_block();
    f32x8 lRow = *(const f32x8*)(myS + hi16 * 8);
    f32x8 inv;
#pragma unroll
    for (int v = 0; v < 8; ++v) inv[v] = 1.f / lRow[v];

#pragma unroll
    for (int dt = 0; dt < 8; ++dt)
#pragma unroll
        for (int v = 0; v < 8; ++v) {
            int row = t0 + v + 8 * hi16;
            int col = h * HIDD + dt * 16 + tc;
            Res[(size_t)row * NCOLS + col] = f2bf(O[dt][v] * inv[v]);
        }
}

// ---------------------------------------------------------------------------
// Kernel 3: output GEMM  out[f32] = Res[bf16, 4096x1024] @ Wo[f32, 1024x512]
// ---------------------------------------------------------------------------
__global__ __launch_bounds__(256)
void out_gemm(const __bf16* __restrict__ Xb, const float* __restrict__ W,
              float* __restrict__ Out) {
    __shared__ __align__(32) __bf16 Xl[128 * 32];
    __shared__ __align__(32) __bf16 Wl[128 * 32];

    const int tid  = threadIdx.x;
    const int lane = tid & 31;
    const int wv   = tid >> 5;
    const int hi16 = lane >> 4;
    const int tc   = lane & 15;
    const int wm   = (wv >> 2) * 64;
    const int wn   = (wv & 3) * 32;
    const int m0   = blockIdx.x * 128;
    const int n0   = blockIdx.y * 128;

    f32x8 acc[4][2] = {};

    for (int k0 = 0; k0 < NCOLS; k0 += 32) {
        // X tile: already bf16, copy as 16B vectors (2 per thread)
#pragma unroll
        for (int i = 0; i < 2; ++i) {
            int lin = tid * 8 + i * 2048;
            int r = lin >> 5, c = lin & 31;
            *(bf16x8*)(Xl + lin) =
                *(const bf16x8*)(Xb + (size_t)(m0 + r) * NCOLS + k0 + c);
        }
        // W tile: f32 -> bf16, stored transposed [n][k]
#pragma unroll
        for (int i = 0; i < 4; ++i) {
            int lin = tid * 4 + i * 1024;
            int wk = lin >> 7, nn = lin & 127;
            f32x4 wvv = *(const f32x4*)(W + (size_t)(k0 + wk) * F_OUT + n0 + nn);
#pragma unroll
            for (int e = 0; e < 4; ++e) Wl[(nn + e) * 32 + wk] = f2bf(wvv[e]);
        }
        __syncthreads();

        bf16x16 bf[2];
#pragma unroll
        for (int j = 0; j < 2; ++j)
            bf[j] = load_afrag(Wl + (wn + j * 16 + tc) * 32 + hi16 * 8);
#pragma unroll
        for (int i = 0; i < 4; ++i) {
            bf16x16 af = load_afrag(Xl + (wm + i * 16 + tc) * 32 + hi16 * 8);
#pragma unroll
            for (int j = 0; j < 2; ++j)
                acc[i][j] = wmma_bf16(af, bf[j], acc[i][j]);
        }
        __syncthreads();
    }

#pragma unroll
    for (int i = 0; i < 4; ++i)
#pragma unroll
        for (int j = 0; j < 2; ++j)
#pragma unroll
            for (int v = 0; v < 8; ++v) {
                int row = m0 + wm + i * 16 + v + 8 * hi16;
                int col = n0 + wn + j * 16 + tc;
                Out[(size_t)row * F_OUT + col] = acc[i][j][v];
            }
}

// ---------------------------------------------------------------------------
extern "C" void kernel_launch(void* const* d_in, const int* in_sizes, int n_in,
                              void* d_out, int out_size, void* d_ws, size_t ws_size,
                              hipStream_t stream) {
    (void)in_sizes; (void)n_in; (void)out_size; (void)ws_size;
    const float* inputs = (const float*)d_in[0];
    const int*   index  = (const int*)d_in[1];
    const float* Wk     = (const float*)d_in[2];
    const float* Wq     = (const float*)d_in[3];
    const float* Wv     = (const float*)d_in[4];
    const float* Wo     = (const float*)d_in[5];
    float* out = (float*)d_out;

    char* ws = (char*)d_ws;
    __bf16* Qbf = (__bf16*)(ws + ((size_t)0 << 20));
    __bf16* Kbf = (__bf16*)(ws + ((size_t)8 << 20));
    __bf16* Vt  = (__bf16*)(ws + ((size_t)16 << 20));
    __bf16* Res = (__bf16*)(ws + ((size_t)24 << 20));

    dim3 gProj(T_TOK / 128, NCOLS / 128);
    proj_gemm<false><<<gProj, 256, 0, stream>>>(inputs, Wq, Qbf);
    proj_gemm<false><<<gProj, 256, 0, stream>>>(inputs, Wk, Kbf);
    proj_gemm<true ><<<gProj, 256, 0, stream>>>(inputs, Wv, Vt);

    attn_kernel<<<(T_TOK / 16) * NHEAD / 4, 128, 0, stream>>>(Qbf, Kbf, Vt, index, Res);

    out_gemm<<<dim3(T_TOK / 128, F_OUT / 128), 256, 0, stream>>>(Res, Wo, out);
}
